// TagProjector_56495999811635
// MI455X (gfx1250) — matrix operand order repaired and checked by reading, not compiled, not run on hardware
//
#include <hip/hip_runtime.h>
#include <hip/hip_bf16.h>

typedef __attribute__((ext_vector_type(16))) _Float16 v16h;
typedef __attribute__((ext_vector_type(8)))  _Float16 v8h;
typedef __attribute__((ext_vector_type(8)))  float    v8f;

#define D_CLIP 512
#define D_HID  1024
#define D_OUT  256
#define NROWS  (1024 * 50)   // B * P
#define TM     32            // rows per workgroup (two WMMA M-tiles)

// ---------------------------------------------------------------------------
// Prep: transpose + convert weights to f16 [N][K] so WMMA B-fragments are
// contiguous 32-byte reads per lane. 1.5 MB total -> L2-resident for the run.
// ---------------------------------------------------------------------------
__global__ void prep_weights(const float* __restrict__ w1,
                             const float* __restrict__ w2,
                             _Float16* __restrict__ w1t,
                             _Float16* __restrict__ w2t) {
    int t = blockIdx.x * blockDim.x + threadIdx.x;
    if (t < D_CLIP * D_HID) {
        int k = t / D_HID, n = t % D_HID;              // w1 is [K=512][N=1024]
        w1t[n * D_CLIP + k] = (_Float16)w1[t];
    } else {
        int u = t - D_CLIP * D_HID;
        if (u < D_HID * D_OUT) {
            int k = u / D_OUT, n = u % D_OUT;          // w2 is [K=1024][N=256]
            w2t[n * D_HID + k] = (_Float16)w2[u];
        }
    }
}

__device__ __forceinline__ float gelu_tanh(float x) {
    // jax.nn.gelu default (approximate=True)
    float u = 0.7978845608028654f * (x + 0.044715f * x * x * x);
    return 0.5f * x * (1.0f + tanhf(u));
}

// A-fragment: lane l -> row l%16, one contiguous 16-half chunk at k offset
// (l>=16 ? 16 : 0), matching the contiguous B-fragment order.
__device__ __forceinline__ v16h load_a_frag(const _Float16* p) {
    return *(const v16h*)p;
}

// ---------------------------------------------------------------------------
// Fused: gather -> GEMM1(+bias,GELU) -> GEMM2(+bias) -> L2 normalize -> mask.
// One workgroup = 32 rows (two M-tiles). 256 threads = 8 waves (wave32).
// Each wave keeps TWO accumulators per N-tile so every B-fragment load feeds
// two independent WMMA chains; B is software-pipelined one k-step ahead.
// LDS: A-tile (32 KB) aliased with Y-tile (32 KB) + H-tile (64 KB) = 96 KB.
// ---------------------------------------------------------------------------
__launch_bounds__(256)
__global__ void tagproj_fused(const float* __restrict__ tab,
                              const _Float16* __restrict__ w1t,
                              const float* __restrict__ b1,
                              const _Float16* __restrict__ w2t,
                              const float* __restrict__ b2,
                              const int* __restrict__ ids,
                              float* __restrict__ out) {
    __shared__ __align__(32) unsigned char smem[TM * D_CLIP * 2 + TM * D_HID * 2];
    _Float16* sA = (_Float16*)smem;                       // 32 KB (stage 0/1)
    float*    sY = (float*)smem;                          // 32 KB (aliases sA)
    _Float16* sH = (_Float16*)(smem + TM * D_CLIP * 2);   // 64 KB
    __shared__ int sId[TM];

    const int tid  = threadIdx.x;
    const int wave = tid >> 5;
    const int lane = tid & 31;
    const int nsel = lane & 15;          // N (and A-row M) index within tile
    const int hi16 = lane >> 4;          // upper half-wave flag
    const int kb   = hi16 * 16;          // A/B K sub-offset (halves)
    const int r0   = blockIdx.x * TM;    // first global row of this tile

    if (tid < TM) sId[tid] = ids[r0 + tid];
    __syncthreads();

    // ---- Stage 0: gather 32 rows x 512 f32, convert to f16 into LDS -------
#pragma unroll
    for (int i = 0; i < (TM * D_CLIP / 4) / 256; ++i) {   // 16 iters
        int idx = tid + 256 * i;
        int row = idx >> 7;              // 128 float4 per row
        int c4  = idx & 127;
        int cid = sId[row]; cid = cid < 0 ? 0 : cid;      // clamp like reference
        const float4 v = ((const float4*)(tab + (size_t)cid * D_CLIP))[c4];
        _Float16* dst = &sA[row * D_CLIP + c4 * 4];
        dst[0] = (_Float16)v.x; dst[1] = (_Float16)v.y;
        dst[2] = (_Float16)v.z; dst[3] = (_Float16)v.w;
    }
    __syncthreads();

    // ---- Stage 1: GEMM1 [32x512]x[512x1024] + bias + GELU -> sH ------------
    // 64 N-tiles, 8 per wave; 2 M-tiles via dual accumulators; K -> 16 steps.
    for (int i = 0; i < 8; ++i) {
        const int nt = wave * 8 + i;
        const int n  = nt * 16 + nsel;
        v8f c0, c1;
        {
            const float bv = b1[n];
#pragma unroll
            for (int j = 0; j < 8; ++j) { c0[j] = bv; c1[j] = bv; }
        }
        const _Float16* brow = w1t + (size_t)n * D_CLIP + hi16 * 16;
        const _Float16* a0   = sA + nsel * D_CLIP + kb;
        const _Float16* a1   = sA + (16 + nsel) * D_CLIP + kb;
        v16h b = *(const v16h*)brow;
#pragma unroll
        for (int ks = 0; ks < D_CLIP / 32; ++ks) {
            const int k0 = ks * 32;
            v16h bn = b;
            if (ks < D_CLIP / 32 - 1) bn = *(const v16h*)(brow + k0 + 32);
            v16h af0 = load_a_frag(a0 + k0);
            v16h af1 = load_a_frag(a1 + k0);
            c0 = __builtin_amdgcn_wmma_f32_16x16x32_f16(
                     false, af0, false, b, (short)0, c0, false, false);
            c1 = __builtin_amdgcn_wmma_f32_16x16x32_f16(
                     false, af1, false, b, (short)0, c1, false, false);
            b = bn;
        }
#pragma unroll
        for (int j = 0; j < 8; ++j) {
            const int m = j + hi16 * 8;
            sH[m * D_HID + n]        = (_Float16)gelu_tanh(c0[j]);
            sH[(m + 16) * D_HID + n] = (_Float16)gelu_tanh(c1[j]);
        }
    }
    __syncthreads();   // sA dead from here on; sY (alias) becomes live below

    // ---- Stage 2: GEMM2 [32x1024]x[1024x256] + bias -> sY ------------------
    // 16 N-tiles, 2 per wave; 2 M-tiles via dual accumulators; K -> 32 steps.
    for (int i = 0; i < 2; ++i) {
        const int nt = wave * 2 + i;
        const int n  = nt * 16 + nsel;
        v8f c0, c1;
        {
            const float bv = b2[n];
#pragma unroll
            for (int j = 0; j < 8; ++j) { c0[j] = bv; c1[j] = bv; }
        }
        const _Float16* brow = w2t + (size_t)n * D_HID + hi16 * 16;
        const _Float16* a0   = sH + nsel * D_HID + kb;
        const _Float16* a1   = sH + (16 + nsel) * D_HID + kb;
        v16h b = *(const v16h*)brow;
#pragma unroll
        for (int ks = 0; ks < D_HID / 32; ++ks) {
            const int k0 = ks * 32;
            v16h bn = b;
            if (ks < D_HID / 32 - 1) bn = *(const v16h*)(brow + k0 + 32);
            v16h af0 = load_a_frag(a0 + k0);
            v16h af1 = load_a_frag(a1 + k0);
            c0 = __builtin_amdgcn_wmma_f32_16x16x32_f16(
                     false, af0, false, b, (short)0, c0, false, false);
            c1 = __builtin_amdgcn_wmma_f32_16x16x32_f16(
                     false, af1, false, b, (short)0, c1, false, false);
            b = bn;
        }
#pragma unroll
        for (int j = 0; j < 8; ++j) {
            const int m = j + hi16 * 8;
            sY[m * D_OUT + n]        = c0[j];
            sY[(m + 16) * D_OUT + n] = c1[j];
        }
    }
    __syncthreads();

    // ---- Stage 3: per-row L2 normalize + pad mask + store ------------------
    // 32 rows, 4 per wave; 8 elements per lane, wave-wide shuffle reduction.
#pragma unroll
    for (int rr = 0; rr < 4; ++rr) {
        const int r = wave * 4 + rr;
        float ss = 0.0f;
#pragma unroll
        for (int j = 0; j < 8; ++j) {
            const float v = sY[r * D_OUT + lane + 32 * j];
            ss += v * v;
        }
#pragma unroll
        for (int off = 16; off > 0; off >>= 1)
            ss += __shfl_xor(ss, off, 32);
        const float inv   = 1.0f / fmaxf(sqrtf(ss), 1e-12f);
        const float scale = (sId[r] < 0) ? 0.0f : inv;   // zero padded rows
        float* orow = out + (size_t)(r0 + r) * D_OUT;
#pragma unroll
        for (int j = 0; j < 8; ++j) {
            const int n = lane + 32 * j;
            orow[n] = sY[r * D_OUT + n] * scale;
        }
    }
}

// ---------------------------------------------------------------------------
extern "C" void kernel_launch(void* const* d_in, const int* in_sizes, int n_in,
                              void* d_out, int out_size, void* d_ws, size_t ws_size,
                              hipStream_t stream) {
    const float* tab = (const float*)d_in[0];   // [100000, 512]
    const float* w1  = (const float*)d_in[1];   // [512, 1024]
    const float* b1  = (const float*)d_in[2];   // [1024]
    const float* w2  = (const float*)d_in[3];   // [1024, 256]
    const float* b2  = (const float*)d_in[4];   // [256]
    const int*   ids = (const int*)d_in[5];     // [1024, 50]
    float* out = (float*)d_out;                 // [1024, 50, 256]

    _Float16* w1t = (_Float16*)d_ws;                                       // 1 MB
    _Float16* w2t = (_Float16*)((char*)d_ws + (size_t)D_HID * D_CLIP * 2); // +0.5 MB

    const int prep_n = D_CLIP * D_HID + D_HID * D_OUT;
    prep_weights<<<(prep_n + 255) / 256, 256, 0, stream>>>(w1, w2, w1t, w2t);

    tagproj_fused<<<NROWS / TM, 256, 0, stream>>>(tab, w1t, b1, w2t, b2, ids, out);
}
